// Mamba3Vision_40235253629432
// MI455X (gfx1250) — matrix operand order repaired and checked by reading, not compiled, not run on hardware
//
#include <hip/hip_runtime.h>

// ---------------- model dims ----------------
#define IMGSZ 384
#define PS 16
#define CINCH 3
#define DIMM 512
#define DEPTH 8
#define NSTATE 32
#define HDIM 64
#define NCLS 2
#define NBATCH 32
#define DINNER 1024          // EXP*DIM
#define NHEADS 16            // DINNER/HDIM
#define CONVD 1088           // DINNER + 2*NSTATE
#define DPROJ 2128           // 2*DINNER + 2*NSTATE + NHEADS
#define LSEQ 576             // (384/16)^2
#define RDIM 16              // NSTATE/2
#define KCONV 4
#define MROWS (NBATCH*LSEQ)  // 18432
#define LNEPS 1e-6f

// ---------------- CDNA5 WMMA types ----------------
typedef __attribute__((ext_vector_type(16))) __bf16 v16bf;
typedef __attribute__((ext_vector_type(8)))  float  v8f;
typedef unsigned int u32x4 __attribute__((ext_vector_type(4)));

union FragBF { v16bf v; u32x4 q[2]; };

static __device__ inline unsigned short f2bf(float x) {
    unsigned int u = __float_as_uint(x);
    unsigned int r = (u + 0x7FFFu + ((u >> 16) & 1u)) >> 16;
    return (unsigned short)r;
}
static __device__ inline float siluf(float x) { return x / (1.f + __expf(-x)); }

// low 32 bits of a generic pointer to LDS == DS byte address (aperture rule)
static __device__ inline unsigned lds_addr(const void* p) {
    return (unsigned)(size_t)p;
}

// ======================================================================
// GEMM: C[M,N] = A[M,K](bf16) * Bt[N,K](bf16)^T (+bias[N]) (+addend[M,N])
// Block 256 thr (8 waves). Block tile 128x64; wave tile 32x32 (2x2 WMMA).
// K staged in 32-chunks DMA'd straight into LDS with
// GLOBAL_LOAD_ASYNC_TO_LDS_B128 (ASYNCcnt), double buffered: chunk k+1's
// DMA flies while chunk k feeds the WMMAs; s_wait_asynccnt <= 2+hasB
// (async loads complete in order) gates consumption of the older chunk.
// M must be multiple of 128, K of 32.
// ======================================================================
#define GT_M 128
#define GT_N 64
#define GT_K 32
#define LDSW 40   // padded row stride (elements) -> 80B rows, 16B aligned segs

__global__ __launch_bounds__(256) void gemm_bf16_wmma(
    const unsigned short* __restrict__ A,
    const unsigned short* __restrict__ Bt,
    const float* __restrict__ bias,
    const float* __restrict__ addend,
    float* __restrict__ C, int M, int N, int K)
{
    __shared__ __align__(16) unsigned short sA[2][GT_M * LDSW];
    __shared__ __align__(16) unsigned short sB[2][GT_N * LDSW];

    const int tid  = threadIdx.x;
    const int lane = tid & 31;
    const int wid  = tid >> 5;
    const int wm   = (wid >> 1) * 32;      // 4 waves along M
    const int wn   = (wid & 1) * 32;       // 2 waves along N
    const int m0   = blockIdx.y * GT_M;
    const int n0   = blockIdx.x * GT_N;
    const int half = lane >> 4;            // K-half selector per ISA layout
    const int lr   = lane & 15;

    v8f acc00 = {0.f,0.f,0.f,0.f,0.f,0.f,0.f,0.f};
    v8f acc01 = acc00, acc10 = acc00, acc11 = acc00;

    const int arow = tid >> 1, aseg = tid & 1;   // 128 rows x 32 elems: 16 bf16/thread
    const int brow = tid >> 2, bseg = tid & 3;   // 64 rows x 32 elems: 8 bf16/thread

    const bool nvalid   = (n0 + brow) < N;            // per-lane B-row guard
    const bool waveHasB = (n0 + (wid << 3)) < N;      // wave-uniform: any lane issues B DMA

    const unsigned lAbuf[2] = { lds_addr(&sA[0][arow * LDSW + aseg * 16]),
                                lds_addr(&sA[1][arow * LDSW + aseg * 16]) };
    const unsigned lBbuf[2] = { lds_addr(&sB[0][brow * LDSW + bseg * 8]),
                                lds_addr(&sB[1][brow * LDSW + bseg * 8]) };

    // issue one K-chunk's DMA into LDS buffer `buf`
    auto issue = [&](int kkv, int buf) {
        unsigned long long gA =
            (unsigned long long)(size_t)(A + (size_t)(m0 + arow) * K + kkv + aseg * 16);
        // single IOFFSET applies to both LDS and global sides -> offset:16
        asm volatile(
            "global_load_async_to_lds_b128 %0, %1, off\n\t"
            "global_load_async_to_lds_b128 %0, %1, off offset:16"
            :: "v"(lAbuf[buf]), "v"(gA) : "memory");
        if (nvalid) {
            unsigned long long gB =
                (unsigned long long)(size_t)(Bt + (size_t)(n0 + brow) * K + kkv + bseg * 8);
            asm volatile("global_load_async_to_lds_b128 %0, %1, off"
                         :: "v"(lBbuf[buf]), "v"(gB) : "memory");
        }
    };

    // prologue: zero-fill OOB B rows once (DMA never writes them), start chunk 0
    if (!nvalid) {
        u32x4 z4 = {0u,0u,0u,0u};
        *(u32x4*)(&sB[0][brow * LDSW + bseg * 8]) = z4;
        *(u32x4*)(&sB[1][brow * LDSW + bseg * 8]) = z4;
    }
    issue(0, 0);

    int ib = 0;
    for (int kk = 0; kk < K; kk += GT_K, ib ^= 1) {
        const bool hasNext = (kk + GT_K) < K;
        if (hasNext) {
            issue(kk + GT_K, ib ^ 1);                 // keep next chunk's DMA in flight
            if (kk + 2 * GT_K < K)                    // warm L2 for the chunk after next
                __builtin_prefetch(A + (size_t)(m0 + arow) * K + kk + 2 * GT_K + aseg * 16, 0, 3);
            // async loads retire in order: <= (next chunk's count) => current chunk done
            if (waveHasB) asm volatile("s_wait_asynccnt 0x3" ::: "memory");
            else          asm volatile("s_wait_asynccnt 0x2" ::: "memory");
        } else {
            asm volatile("s_wait_asynccnt 0x0" ::: "memory");
        }
        __syncthreads();

        // ---- fragments from buffer ib ----
        const unsigned short* cA = sA[ib];
        const unsigned short* cB = sB[ib];
        // A 16x32 bf16 layout: lane(l<16)=row l: K 0..7 & 16..23 ; lane l+16: K 8..15 & 24..31
        FragBF a0, a1, b0, b1;
        {
            const unsigned short* r = cA + (wm + lr) * LDSW;
            a0.q[0] = *(const u32x4*)(r + half * 8);
            a0.q[1] = *(const u32x4*)(r + 16 + half * 8);
            const unsigned short* r2 = cA + (wm + 16 + lr) * LDSW;
            a1.q[0] = *(const u32x4*)(r2 + half * 8);
            a1.q[1] = *(const u32x4*)(r2 + 16 + half * 8);
        }
        // B 32x16 bf16: lanes 0-15 hold K=0..7 & 16..23 of col=lane, lanes 16-31 K=8..15 & 24..31
        {
            const unsigned short* r = cB + (wn + lr) * LDSW + half * 8;
            b0.q[0] = *(const u32x4*)(r);
            b0.q[1] = *(const u32x4*)(r + 16);
            const unsigned short* r2 = cB + (wn + 16 + lr) * LDSW + half * 8;
            b1.q[0] = *(const u32x4*)(r2);
            b1.q[1] = *(const u32x4*)(r2 + 16);
        }

        acc00 = __builtin_amdgcn_wmma_f32_16x16x32_bf16(false, a0.v, false, b0.v, (short)0, acc00, false, false);
        acc01 = __builtin_amdgcn_wmma_f32_16x16x32_bf16(false, a0.v, false, b1.v, (short)0, acc01, false, false);
        acc10 = __builtin_amdgcn_wmma_f32_16x16x32_bf16(false, a1.v, false, b0.v, (short)0, acc10, false, false);
        acc11 = __builtin_amdgcn_wmma_f32_16x16x32_bf16(false, a1.v, false, b1.v, (short)0, acc11, false, false);

        // all waves done reading buffer ib before it is re-targeted next iteration
        __syncthreads();
    }

    // ---- epilogue: C 16x16 f32 layout: lanes0-15 vgpr r -> M=r, lanes16-31 -> M=r+8; N=lane&15
#define EMIT(ACC, MI, NI)                                                        \
    {                                                                            \
        int gr = m0 + wm + (MI) + rr + 8 * half;                                 \
        int gc = n0 + wn + (NI) + lr;                                            \
        if (gr < M && gc < N) {                                                  \
            float vv = (ACC)[rr];                                                \
            if (bias)   vv += bias[gc];                                          \
            if (addend) vv += addend[(size_t)gr * N + gc];                       \
            C[(size_t)gr * N + gc] = vv;                                         \
        }                                                                        \
    }
#pragma unroll
    for (int rr = 0; rr < 8; ++rr) {
        EMIT(acc00, 0,  0)
        EMIT(acc01, 0,  16)
        EMIT(acc10, 16, 0)
        EMIT(acc11, 16, 16)
    }
#undef EMIT
}

// ======================================================================
// weight conversion kernels
// ======================================================================
__global__ void k_f32_to_bf16(const float* __restrict__ in, unsigned short* __restrict__ out, long n) {
    long i = (long)blockIdx.x * blockDim.x + threadIdx.x;
    if (i < n) out[i] = f2bf(in[i]);
}
// in: [layers][K][N] f32 -> out: [layers][N][K] bf16
__global__ void k_transpose_bf16(const float* __restrict__ in, unsigned short* __restrict__ out,
                                 int K, int N, long total) {
    long i = (long)blockIdx.x * blockDim.x + threadIdx.x;
    if (i >= total) return;
    long per = (long)K * N;
    long layer = i / per, rem = i % per;
    long n = rem / K, k = rem % K;
    out[i] = f2bf(in[layer * per + k * N + n]);
}

// ======================================================================
// im2col for patch embed: A[b*L+l][c*256+py*16+px]
// ======================================================================
__global__ void k_im2col(const float* __restrict__ x, unsigned short* __restrict__ out, long total) {
    long i = (long)blockIdx.x * blockDim.x + threadIdx.x;
    if (i >= total) return;
    int k = (int)(i % (CINCH * PS * PS));
    long m = i / (CINCH * PS * PS);
    int b = (int)(m / LSEQ), l = (int)(m % LSEQ);
    int ty = l / (IMGSZ / PS), tx = l % (IMGSZ / PS);
    int c = k / (PS * PS), r = k % (PS * PS);
    int py = r / PS, px = r % PS;
    long xi = (((long)b * CINCH + c) * IMGSZ + ty * PS + py) * IMGSZ + tx * PS + px;
    out[i] = f2bf(x[xi]);
}

__global__ void k_pos_add(float* __restrict__ t, const float* __restrict__ pos, long total) {
    long i = (long)blockIdx.x * blockDim.x + threadIdx.x;
    if (i >= total) return;
    long row = i / DIMM;
    int d = (int)(i % DIMM);
    int l = (int)(row % LSEQ);
    t[i] += pos[(long)l * DIMM + d];
}

// ======================================================================
// row LayerNorm over 512, writes bf16 and/or f32
// ======================================================================
__global__ __launch_bounds__(256) void k_ln512(const float* __restrict__ X,
        const float* __restrict__ w, const float* __restrict__ bsh,
        unsigned short* __restrict__ out_bf, float* __restrict__ out_f)
{
    int row = blockIdx.x, tid = threadIdx.x;
    const float* xr = X + (size_t)row * DIMM;
    float v0 = xr[tid], v1 = xr[tid + 256];
    __shared__ float red[256];
    red[tid] = v0 + v1; __syncthreads();
    for (int s = 128; s > 0; s >>= 1) { if (tid < s) red[tid] += red[tid + s]; __syncthreads(); }
    float m = red[0] * (1.f / DIMM);
    __syncthreads();
    float d0 = v0 - m, d1 = v1 - m;
    red[tid] = d0 * d0 + d1 * d1; __syncthreads();
    for (int s = 128; s > 0; s >>= 1) { if (tid < s) red[tid] += red[tid + s]; __syncthreads(); }
    float r = rsqrtf(red[0] * (1.f / DIMM) + LNEPS);
    float o0 = d0 * r * w[tid] + bsh[tid];
    float o1 = d1 * r * w[tid + 256] + bsh[tid + 256];
    if (out_bf) {
        out_bf[(size_t)row * DIMM + tid] = f2bf(o0);
        out_bf[(size_t)row * DIMM + 256 + tid] = f2bf(o1);
    }
    if (out_f) {
        out_f[(size_t)row * DIMM + tid] = o0;
        out_f[(size_t)row * DIMM + 256 + tid] = o1;
    }
}

// ======================================================================
// depthwise causal conv (K=4) + bias + SiLU over xBC slice of zxbcdt
// ======================================================================
__global__ void k_conv_silu(const float* __restrict__ zx, const float* __restrict__ cw,
                            const float* __restrict__ cb, float* __restrict__ xact, long total)
{
    long i = (long)blockIdx.x * blockDim.x + threadIdx.x;
    if (i >= total) return;
    int c = (int)(i % CONVD);
    long bl = i / CONVD;
    int l = (int)(bl % LSEQ);
    long b = bl / LSEQ;
    float s = cb[c];
#pragma unroll
    for (int j = 0; j < KCONV; ++j) {
        int ls = l - (KCONV - 1) + j;
        if (ls >= 0)
            s += cw[c * KCONV + j] * zx[((size_t)(b * LSEQ + ls)) * DPROJ + DINNER + c];
    }
    xact[i] = siluf(s);
}

// ======================================================================
// RoPE(B,C) + softplus(dt)+decay, one thread per (b,l)
// ======================================================================
__global__ void k_prep(const float* __restrict__ zx, const float* __restrict__ xact,
                       const float* __restrict__ Bb, const float* __restrict__ Cb,
                       const float* __restrict__ dtb, const float* __restrict__ Alog,
                       float* __restrict__ Bm, float* __restrict__ Cm,
                       float* __restrict__ dts, float* __restrict__ dec, long total)
{
    long bl = (long)blockIdx.x * blockDim.x + threadIdx.x;
    if (bl >= total) return;
    int l = (int)(bl % LSEQ);
    const float* xb = xact + bl * CONVD + DINNER;
#pragma unroll
    for (int i = 0; i < 8; ++i) {
        float inv = __expf(-1.1512925465f * (float)i);   // 10000^(-2i/16)
        float ang = (float)l * inv;
        float sn, cs; __sincosf(ang, &sn, &cs);
        float b1 = xb[i] + Bb[i], b2 = xb[i + 8] + Bb[i + 8];
        Bm[bl * NSTATE + i]     = b1 * cs - b2 * sn;
        Bm[bl * NSTATE + 8 + i] = b1 * sn + b2 * cs;
        float c1 = xb[NSTATE + i] + Cb[i], c2 = xb[NSTATE + i + 8] + Cb[i + 8];
        Cm[bl * NSTATE + i]     = c1 * cs - c2 * sn;
        Cm[bl * NSTATE + 8 + i] = c1 * sn + c2 * cs;
    }
#pragma unroll
    for (int i = 16; i < NSTATE; ++i) {
        Bm[bl * NSTATE + i] = xb[i] + Bb[i];
        Cm[bl * NSTATE + i] = xb[NSTATE + i] + Cb[i];
    }
#pragma unroll
    for (int h = 0; h < NHEADS; ++h) {
        float raw = zx[bl * DPROJ + DINNER + CONVD + h] + dtb[h];
        float sp = (raw > 20.f) ? raw : log1pf(__expf(raw));
        dts[bl * NHEADS + h] = sp;
        dec[bl * NHEADS + h] = __expf(-sp * __expf(Alog[h]));
    }
}

// ======================================================================
// selective scan: one block per (b,h), 64 threads (=HD lanes),
// 32-state per thread in registers, sequential over L=576
// ======================================================================
__global__ __launch_bounds__(64) void k_scan(const float* __restrict__ xact,
        const float* __restrict__ Bm, const float* __restrict__ Cm,
        const float* __restrict__ dts, const float* __restrict__ dec,
        const float* __restrict__ Dsk, float* __restrict__ y)
{
    int b = blockIdx.x / NHEADS;
    int hh = blockIdx.x % NHEADS;
    int p = threadIdx.x;
    float hst[NSTATE];
#pragma unroll
    for (int n = 0; n < NSTATE; ++n) hst[n] = 0.f;
    float dsk = Dsk[hh];
    __shared__ float sB[NSTATE], sC[NSTATE];

    for (int l = 0; l < LSEQ; ++l) {
        long base = (long)b * LSEQ + l;
        if (p < NSTATE) sB[p] = Bm[base * NSTATE + p];
        else            sC[p - NSTATE] = Cm[base * NSTATE + (p - NSTATE)];
        __syncthreads();
        float xv = xact[base * CONVD + hh * HDIM + p];
        float d  = dts[base * NHEADS + hh];
        float a  = dec[base * NHEADS + hh];
        float dx = d * xv;
        float accv = 0.f;
#pragma unroll
        for (int n = 0; n < NSTATE; ++n) {
            hst[n] = a * hst[n] + dx * sB[n];
            accv += hst[n] * sC[n];
        }
        y[base * DINNER + hh * HDIM + p] = accv + dsk * xv;
        __syncthreads();
    }
}

// ======================================================================
// gate with silu(z), RMSNorm with gn_w, write bf16 rows of 1024
// ======================================================================
__global__ __launch_bounds__(256) void k_gate_rms(const float* __restrict__ y,
        const float* __restrict__ zx, const float* __restrict__ gn,
        unsigned short* __restrict__ out)
{
    long row = blockIdx.x;
    int tid = threadIdx.x;
    float v[4]; float ss = 0.f;
#pragma unroll
    for (int j = 0; j < 4; ++j) {
        int c = tid + j * 256;
        float g = y[row * DINNER + c] * siluf(zx[row * DPROJ + c]);
        v[j] = g; ss += g * g;
    }
    __shared__ float red[256];
    red[tid] = ss; __syncthreads();
    for (int s = 128; s > 0; s >>= 1) { if (tid < s) red[tid] += red[tid + s]; __syncthreads(); }
    float r = rsqrtf(red[0] * (1.f / DINNER) + LNEPS);
#pragma unroll
    for (int j = 0; j < 4; ++j) {
        int c = tid + j * 256;
        out[row * DINNER + c] = f2bf(v[j] * r * gn[c]);
    }
}

// ======================================================================
// final: mean over L of LN'ed features, then head [512 -> 2]
// ======================================================================
__global__ __launch_bounds__(256) void k_mean_head(const float* __restrict__ hf,
        const float* __restrict__ hw, const float* __restrict__ hb, float* __restrict__ out)
{
    int b = blockIdx.x, tid = threadIdx.x;
    float s0 = 0.f, s1 = 0.f;
    for (int l = 0; l < LSEQ; ++l) {
        const float* r = hf + ((size_t)b * LSEQ + l) * DIMM;
        s0 += r[tid]; s1 += r[tid + 256];
    }
    __shared__ float sm[DIMM];
    sm[tid] = s0 * (1.f / LSEQ);
    sm[tid + 256] = s1 * (1.f / LSEQ);
    __syncthreads();
    if (tid < NCLS) {
        float acc = hb[tid];
        for (int d = 0; d < DIMM; ++d) acc += sm[d] * hw[d * NCLS + tid];
        out[b * NCLS + tid] = acc;
    }
}

// ======================================================================
// host orchestration
// ======================================================================
extern "C" void kernel_launch(void* const* d_in, const int* in_sizes, int n_in,
                              void* d_out, int out_size, void* d_ws, size_t ws_size,
                              hipStream_t stream)
{
    const float* x       = (const float*)d_in[0];
    const float* patch_w = (const float*)d_in[1];
    const float* patch_b = (const float*)d_in[2];
    const float* pos     = (const float*)d_in[3];
    const float* ln_w    = (const float*)d_in[4];
    const float* ln_b    = (const float*)d_in[5];
    const float* in_proj = (const float*)d_in[6];
    const float* conv_w  = (const float*)d_in[7];
    const float* conv_b  = (const float*)d_in[8];
    const float* dt_bias = (const float*)d_in[9];
    const float* A_log   = (const float*)d_in[10];
    const float* D_skip  = (const float*)d_in[11];
    const float* B_bias  = (const float*)d_in[12];
    const float* C_bias  = (const float*)d_in[13];
    const float* gn_w    = (const float*)d_in[14];
    const float* out_prj = (const float*)d_in[15];
    const float* norm_w  = (const float*)d_in[16];
    const float* norm_b  = (const float*)d_in[17];
    const float* head_w  = (const float*)d_in[18];
    const float* head_b  = (const float*)d_in[19];
    float* out = (float*)d_out;

    char* ws = (char*)d_ws;
    size_t off = 0;
    auto carve = [&](size_t bytes) -> char* {
        char* p = ws + off;
        off += (bytes + 255) & ~(size_t)255;
        return p;
    };

    const int PK = CINCH * PS * PS;            // 768
    float*          t_res  = (float*)         carve((size_t)MROWS * DIMM * 4);
    unsigned short* a_im   = (unsigned short*)carve((size_t)MROWS * PK * 2);
    unsigned short* w_pat  = (unsigned short*)carve((size_t)DIMM * PK * 2);
    unsigned short* w_in   = (unsigned short*)carve((size_t)DEPTH * DPROJ * DIMM * 2);
    unsigned short* w_out  = (unsigned short*)carve((size_t)DEPTH * DIMM * DINNER * 2);
    unsigned short* h_bf   = (unsigned short*)carve((size_t)MROWS * DIMM * 2);
    float*          zx     = (float*)         carve((size_t)MROWS * DPROJ * 4);
    float*          xact   = (float*)         carve((size_t)MROWS * CONVD * 4);
    float*          Bm     = (float*)         carve((size_t)MROWS * NSTATE * 4);
    float*          Cm     = (float*)         carve((size_t)MROWS * NSTATE * 4);
    float*          dts    = (float*)         carve((size_t)MROWS * NHEADS * 4);
    float*          decb   = (float*)         carve((size_t)MROWS * NHEADS * 4);
    float*          yscan  = (float*)         carve((size_t)MROWS * DINNER * 4);
    unsigned short* y_bf   = (unsigned short*)carve((size_t)MROWS * DINNER * 2);
    float*          hf     = (float*)         carve((size_t)MROWS * DIMM * 4);
    (void)ws_size; (void)in_sizes; (void)n_in; (void)out_size;

    auto nblk = [](long n, int bs) { return (unsigned)((n + bs - 1) / bs); };

    // ---- weight prep (bf16, [N,K] layout) ----
    {
        long n = (long)DIMM * PK;
        k_f32_to_bf16<<<nblk(n, 256), 256, 0, stream>>>(patch_w, w_pat, n);
        long ni = (long)DEPTH * DIMM * DPROJ;
        k_transpose_bf16<<<nblk(ni, 256), 256, 0, stream>>>(in_proj, w_in, DIMM, DPROJ, ni);
        long no = (long)DEPTH * DINNER * DIMM;
        k_transpose_bf16<<<nblk(no, 256), 256, 0, stream>>>(out_prj, w_out, DINNER, DIMM, no);
    }

    // ---- patch embed: im2col + WMMA GEMM + pos ----
    {
        long n = (long)MROWS * PK;
        k_im2col<<<nblk(n, 256), 256, 0, stream>>>(x, a_im, n);
        dim3 g(DIMM / GT_N, MROWS / GT_M);
        gemm_bf16_wmma<<<g, 256, 0, stream>>>(a_im, w_pat, patch_b, nullptr, t_res,
                                              MROWS, DIMM, PK);
        long np = (long)MROWS * DIMM;
        k_pos_add<<<nblk(np, 256), 256, 0, stream>>>(t_res, pos, np);
    }

    // ---- Mamba blocks ----
    for (int layer = 0; layer < DEPTH; ++layer) {
        k_ln512<<<MROWS, 256, 0, stream>>>(t_res, ln_w + layer * DIMM, ln_b + layer * DIMM,
                                           h_bf, nullptr);
        {
            dim3 g((DPROJ + GT_N - 1) / GT_N, MROWS / GT_M);
            gemm_bf16_wmma<<<g, 256, 0, stream>>>(h_bf, w_in + (size_t)layer * DPROJ * DIMM,
                                                  nullptr, nullptr, zx, MROWS, DPROJ, DIMM);
        }
        {
            long n = (long)MROWS * CONVD;
            k_conv_silu<<<nblk(n, 256), 256, 0, stream>>>(zx, conv_w + (size_t)layer * CONVD * KCONV,
                                                          conv_b + layer * CONVD, xact, n);
        }
        k_prep<<<nblk(MROWS, 256), 256, 0, stream>>>(zx, xact,
                B_bias + layer * NSTATE, C_bias + layer * NSTATE,
                dt_bias + layer * NHEADS, A_log + layer * NHEADS,
                Bm, Cm, dts, decb, MROWS);
        k_scan<<<NBATCH * NHEADS, 64, 0, stream>>>(xact, Bm, Cm, dts, decb,
                                                   D_skip + layer * NHEADS, yscan);
        k_gate_rms<<<MROWS, 256, 0, stream>>>(yscan, zx, gn_w + layer * DINNER, y_bf);
        {
            dim3 g(DIMM / GT_N, MROWS / GT_M);
            gemm_bf16_wmma<<<g, 256, 0, stream>>>(y_bf, w_out + (size_t)layer * DIMM * DINNER,
                                                  nullptr, t_res, t_res, MROWS, DIMM, DINNER);
        }
    }

    // ---- final LN + mean + head ----
    k_ln512<<<MROWS, 256, 0, stream>>>(t_res, norm_w, norm_b, nullptr, hf);
    k_mean_head<<<NBATCH, 256, 0, stream>>>(hf, head_w, head_b, out);
}
